// MHABlock_14972255994038
// MI455X (gfx1250) — compile-verified, matrix operand order checked by reading, not run on hardware
//
#include <hip/hip_runtime.h>
#include <hip/hip_bf16.h>

// ---------------------------------------------------------------------------
// MHA block for MI455X (gfx1250, wave32, WMMA bf16 16x16x32).
//   x[4096,768] fp32 -> out[4096,768] fp32
// Pipeline: cvt(f32->bf16) -> QKV GEMM (WMMA, 32x64 reg-blocked) ->
//           flash attention (WMMA, 32q x 32k blocked) ->
//           out projection (WMMA, 32x64 reg-blocked) + bias.
// ---------------------------------------------------------------------------

typedef __bf16 bf16_t;
typedef __attribute__((ext_vector_type(16))) __bf16 v16bf;
typedef __attribute__((ext_vector_type(8)))  __bf16 v8bf;
typedef __attribute__((ext_vector_type(8)))  float  v8f;

#define SEQ   4096
#define EMB   768
#define HEADS 12
#define HD    64
#define QKV_E (3 * EMB)     // 2304

__device__ __forceinline__ v8f wmma_bf16(v16bf a, v16bf b, v8f c) {
    return __builtin_amdgcn_wmma_f32_16x16x32_bf16(
        /*neg_a=*/false, a, /*neg_b=*/false, b,
        /*c_mod=*/(short)0, c, /*reuse_a=*/false, /*reuse_b=*/false);
}

// Load a 16x32 bf16 A-fragment (or symmetric Bt-fragment: lane holds one row
// of a row-major, K-contiguous matrix). ISA 7.12.2 16-bit A 16x32 layout:
// lane = c*16 + r holds row r; elems 0..7 -> K = c*8+0..7, elems 8..15 ->
// K = 16+c*8+0..7. p = tile origin; ld in elements (multiple of 8).
__device__ __forceinline__ v16bf load_frag(const bf16_t* p, int ld) {
    const int lane = threadIdx.x & 31;
    const int c = lane >> 4;
    const int r = lane & 15;
    const bf16_t* base = p + r * ld + c * 8;
    v8bf lo = *(const v8bf*)(base);
    v8bf hi = *(const v8bf*)(base + 16);
    return __builtin_shufflevector(lo, hi, 0, 1, 2, 3, 4, 5, 6, 7,
                                           8, 9, 10, 11, 12, 13, 14, 15);
}

// ---------------------------------------------------------------------------
__global__ void cvt_f32_bf16(const float* __restrict__ in,
                             bf16_t* __restrict__ out, int n) {
    int i = blockIdx.x * blockDim.x + threadIdx.x;
    if (i < n) out[i] = (bf16_t)in[i];
}

// ---------------------------------------------------------------------------
// QKV GEMM: qkv[l,e] = sum_d x[l,d]*w[e,d]. One wave per 32x64 output tile:
// 2 A-frags + 4 B-frags -> 8 WMMA per 32-wide K step (1.5 b128 loads/WMMA).
// Q scaled by sqrt(HD)=8 (reference multiplies scores by INV_SCALE).
// V stored transposed per head: Vt[h][d][k].
__global__ void qkv_gemm(const bf16_t* __restrict__ xh,
                         const bf16_t* __restrict__ wh,
                         bf16_t* __restrict__ Qh,
                         bf16_t* __restrict__ Kh,
                         bf16_t* __restrict__ Vth) {
    const int wave = (blockIdx.x * blockDim.x + threadIdx.x) >> 5;
    const int ET = QKV_E / 64;                 // 36 tiles along e
    const int lt = wave / ET;                  // 0..127
    const int et = wave - lt * ET;
    if (lt >= SEQ / 32) return;
    const int l0 = lt * 32, e0 = et * 64;

    v8f acc[2][4];
#pragma unroll
    for (int i = 0; i < 2; ++i)
#pragma unroll
        for (int j = 0; j < 4; ++j) acc[i][j] = (v8f){};

    for (int k0 = 0; k0 < EMB; k0 += 32) {
        v16bf a0 = load_frag(xh + l0 * EMB + k0, EMB);
        v16bf a1 = load_frag(xh + (l0 + 16) * EMB + k0, EMB);
#pragma unroll
        for (int j = 0; j < 4; ++j) {
            v16bf b = load_frag(wh + (e0 + 16 * j) * EMB + k0, EMB);
            acc[0][j] = wmma_bf16(a0, b, acc[0][j]);
            acc[1][j] = wmma_bf16(a1, b, acc[1][j]);
        }
    }

    // C layout: lane = c*16+n holds column n; VGPR r holds row m = c*8+r.
    const int lane = threadIdx.x & 31;
    const int n = lane & 15, c = lane >> 4;
    const int sec = e0 / EMB;                  // uniform: 64 | 768
#pragma unroll
    for (int i = 0; i < 2; ++i) {
#pragma unroll
        for (int j = 0; j < 4; ++j) {
            const int e = e0 + 16 * j + n;
            const int h = (e - sec * EMB) / HD;
            const int d = e & (HD - 1);
#pragma unroll
            for (int r = 0; r < 8; ++r) {
                const int l = l0 + 16 * i + c * 8 + r;
                const float v = acc[i][j][r];
                if (sec == 0)      Qh[(h * SEQ + l) * HD + d] = (bf16_t)(v * 8.0f);
                else if (sec == 1) Kh[(h * SEQ + l) * HD + d] = (bf16_t)v;
                else               Vth[(h * HD + d) * SEQ + l] = (bf16_t)v;
            }
        }
    }
}

// ---------------------------------------------------------------------------
// Flash attention: one wave per (head, 32-query tile). Per 32-key iteration:
// 4 K-frags feed 8 score WMMAs (shared across 2 q-groups), online softmax on
// two 16-row groups, P via LDS (32x32 bf16), then each of 4 V-frags feeds
// 2 PV WMMAs -> 16 WMMA / iteration.
__global__ void attn(const bf16_t* __restrict__ Qh,
                     const bf16_t* __restrict__ Kh,
                     const bf16_t* __restrict__ Vth,
                     bf16_t* __restrict__ ctxh) {
    __shared__ bf16_t plds[8 * 32 * 32];       // 8 waves/block * 32x32 tile
    const int wib  = threadIdx.x >> 5;
    const int wave = blockIdx.x * (blockDim.x >> 5) + wib;
    const int head = wave >> 7;                // 128 q-tiles per head
    const int q0   = (wave & 127) << 5;
    const int lane = threadIdx.x & 31;
    const int c = lane >> 4, nn = lane & 15;
    bf16_t* myp = plds + wib * (32 * 32);

    const bf16_t* Qbase = Qh + (head * SEQ + q0) * HD;
    v16bf qa[2][2];                            // [q-group][k-step]
    qa[0][0] = load_frag(Qbase, HD);
    qa[0][1] = load_frag(Qbase + 32, HD);
    qa[1][0] = load_frag(Qbase + 16 * HD, HD);
    qa[1][1] = load_frag(Qbase + 16 * HD + 32, HD);

    float m_i[2][8], l_i[2][8];
#pragma unroll
    for (int g = 0; g < 2; ++g)
#pragma unroll
        for (int r = 0; r < 8; ++r) { m_i[g][r] = -1e30f; l_i[g][r] = 0.0f; }
    v8f ctx[2][4];
#pragma unroll
    for (int g = 0; g < 2; ++g)
#pragma unroll
        for (int j = 0; j < 4; ++j) ctx[g][j] = (v8f){};

    const bf16_t* Vbase = Vth + (head * HD) * SEQ;

    for (int k0 = 0; k0 < SEQ; k0 += 32) {
        const bf16_t* Kb = Kh + (head * SEQ + k0) * HD;
        v16bf kb[2][2];                        // [key-subtile][k-step]
        kb[0][0] = load_frag(Kb, HD);
        kb[0][1] = load_frag(Kb + 32, HD);
        kb[1][0] = load_frag(Kb + 16 * HD, HD);
        kb[1][1] = load_frag(Kb + 16 * HD + 32, HD);

        v8f s[2][2];                           // [q-group][key-subtile]
#pragma unroll
        for (int g = 0; g < 2; ++g)
#pragma unroll
            for (int kt = 0; kt < 2; ++kt) {
                v8f z = {};
                v8f t = wmma_bf16(qa[g][0], kb[kt][0], z);
                s[g][kt] = wmma_bf16(qa[g][1], kb[kt][1], t);
            }

        // Online softmax per q-group. Row m = c*8+r is spread over 16 lanes
        // sharing c; xor-shuffles with mask<16 stay in-group on wave32.
#pragma unroll
        for (int g = 0; g < 2; ++g) {
            float alpha[8];
#pragma unroll
            for (int r = 0; r < 8; ++r) {
                float mloc = fmaxf(s[g][0][r], s[g][1][r]);
#pragma unroll
                for (int off = 1; off < 16; off <<= 1)
                    mloc = fmaxf(mloc, __shfl_xor(mloc, off, 32));
                const float mnew = fmaxf(m_i[g][r], mloc);
                alpha[r] = __expf(m_i[g][r] - mnew);
                m_i[g][r] = mnew;
                const float p0 = __expf(s[g][0][r] - mnew);
                const float p1 = __expf(s[g][1][r] - mnew);
                s[g][0][r] = p0; s[g][1][r] = p1;
                float rs = p0 + p1;
#pragma unroll
                for (int off = 1; off < 16; off <<= 1)
                    rs += __shfl_xor(rs, off, 32);
                l_i[g][r] = l_i[g][r] * alpha[r] + rs;
            }
#pragma unroll
            for (int j = 0; j < 4; ++j)
#pragma unroll
                for (int r = 0; r < 8; ++r) ctx[g][j][r] *= alpha[r];

            // P: C-layout f32 -> row-major bf16 rows [g*16 .. g*16+15] in LDS.
#pragma unroll
            for (int r = 0; r < 8; ++r) {
                const int m = g * 16 + c * 8 + r;
                myp[m * 32 + nn]      = (bf16_t)s[g][0][r];
                myp[m * 32 + nn + 16] = (bf16_t)s[g][1][r];
            }
        }
        asm volatile("s_wait_dscnt 0" ::: "memory");
        v16bf pa0 = load_frag(myp, 32);
        v16bf pa1 = load_frag(myp + 16 * 32, 32);

#pragma unroll
        for (int j = 0; j < 4; ++j) {
            v16bf vb = load_frag(Vbase + (j * 16) * SEQ + k0, SEQ);
            ctx[0][j] = wmma_bf16(pa0, vb, ctx[0][j]);
            ctx[1][j] = wmma_bf16(pa1, vb, ctx[1][j]);
        }
    }

    // Normalize and store ctx -> [l][h*64 + d] bf16.
#pragma unroll
    for (int g = 0; g < 2; ++g)
#pragma unroll
        for (int r = 0; r < 8; ++r) {
            const float inv = 1.0f / l_i[g][r];
            const int l = q0 + g * 16 + c * 8 + r;
            bf16_t* o = ctxh + l * EMB + head * HD;
            o[nn]      = (bf16_t)(ctx[g][0][r] * inv);
            o[16 + nn] = (bf16_t)(ctx[g][1][r] * inv);
            o[32 + nn] = (bf16_t)(ctx[g][2][r] * inv);
            o[48 + nn] = (bf16_t)(ctx[g][3][r] * inv);
        }
}

// ---------------------------------------------------------------------------
// Output projection: out[l,e] = sum_d ctx[l,d]*out_w[e,d] + b[e], fp32 out.
// Same 32x64 register blocking as qkv_gemm.
__global__ void out_gemm(const bf16_t* __restrict__ ctxh,
                         const bf16_t* __restrict__ wh,
                         const float* __restrict__ bias,
                         float* __restrict__ out) {
    const int wave = (blockIdx.x * blockDim.x + threadIdx.x) >> 5;
    const int ET = EMB / 64;                   // 12
    const int lt = wave / ET;
    const int et = wave - lt * ET;
    if (lt >= SEQ / 32) return;
    const int l0 = lt * 32, e0 = et * 64;

    v8f acc[2][4];
#pragma unroll
    for (int i = 0; i < 2; ++i)
#pragma unroll
        for (int j = 0; j < 4; ++j) acc[i][j] = (v8f){};

    for (int k0 = 0; k0 < EMB; k0 += 32) {
        v16bf a0 = load_frag(ctxh + l0 * EMB + k0, EMB);
        v16bf a1 = load_frag(ctxh + (l0 + 16) * EMB + k0, EMB);
#pragma unroll
        for (int j = 0; j < 4; ++j) {
            v16bf b = load_frag(wh + (e0 + 16 * j) * EMB + k0, EMB);
            acc[0][j] = wmma_bf16(a0, b, acc[0][j]);
            acc[1][j] = wmma_bf16(a1, b, acc[1][j]);
        }
    }

    const int lane = threadIdx.x & 31;
    const int n = lane & 15, c = lane >> 4;
#pragma unroll
    for (int i = 0; i < 2; ++i) {
#pragma unroll
        for (int j = 0; j < 4; ++j) {
            const int e = e0 + 16 * j + n;
            const float bv = bias[e];
#pragma unroll
            for (int r = 0; r < 8; ++r) {
                const int l = l0 + 16 * i + c * 8 + r;
                out[l * EMB + e] = acc[i][j][r] + bv;
            }
        }
    }
}

// ---------------------------------------------------------------------------
extern "C" void kernel_launch(void* const* d_in, const int* in_sizes, int n_in,
                              void* d_out, int out_size, void* d_ws, size_t ws_size,
                              hipStream_t stream) {
    const float* x     = (const float*)d_in[0];   // [1,4096,768]
    const float* qkv_w = (const float*)d_in[1];   // [2304,768]
    const float* out_w = (const float*)d_in[2];   // [768,768]
    const float* out_b = (const float*)d_in[3];   // [768]
    float* out = (float*)d_out;                   // [1,4096,768]

    char* ws = (char*)d_ws;
    size_t off = 0;
    auto carve = [&](size_t bytes) {
        char* p = ws + off;
        off += (bytes + 255) & ~size_t(255);
        return p;
    };
    bf16_t* xh    = (bf16_t*)carve((size_t)SEQ * EMB * 2);
    bf16_t* qwh   = (bf16_t*)carve((size_t)QKV_E * EMB * 2);
    bf16_t* owh   = (bf16_t*)carve((size_t)EMB * EMB * 2);
    bf16_t* Qh    = (bf16_t*)carve((size_t)HEADS * SEQ * HD * 2);
    bf16_t* Kh    = (bf16_t*)carve((size_t)HEADS * SEQ * HD * 2);
    bf16_t* Vth   = (bf16_t*)carve((size_t)HEADS * HD * SEQ * 2);
    bf16_t* ctxh  = (bf16_t*)carve((size_t)SEQ * EMB * 2);
    (void)ws_size;

    // 1) fp32 -> bf16 conversions
    {
        int n0 = SEQ * EMB, n1 = QKV_E * EMB, n2 = EMB * EMB;
        cvt_f32_bf16<<<(n0 + 255) / 256, 256, 0, stream>>>(x, xh, n0);
        cvt_f32_bf16<<<(n1 + 255) / 256, 256, 0, stream>>>(qkv_w, qwh, n1);
        cvt_f32_bf16<<<(n2 + 255) / 256, 256, 0, stream>>>(out_w, owh, n2);
    }

    // 2) QKV GEMM: (4096/32) * (2304/64) = 4608 waves, 8 waves / block
    {
        int waves = (SEQ / 32) * (QKV_E / 64);
        qkv_gemm<<<waves / 8, 256, 0, stream>>>(xh, qwh, Qh, Kh, Vth);
    }

    // 3) Flash attention: 12 heads * 128 q-tiles = 1536 waves
    {
        int waves = HEADS * (SEQ / 32);
        attn<<<waves / 8, 256, 0, stream>>>(Qh, Kh, Vth, ctxh);
    }

    // 4) Output projection: (4096/32) * (768/64) = 1536 waves
    {
        int waves = (SEQ / 32) * (EMB / 64);
        out_gemm<<<waves / 8, 256, 0, stream>>>(ctxh, owh, out_b, out);
    }
}
